// ConvLayer_7275674599869
// MI455X (gfx1250) — matrix operand order repaired and checked by reading, not compiled, not run on hardware
//
#include <hip/hip_runtime.h>
#include <cstdint>

typedef __attribute__((ext_vector_type(16))) _Float16 v16h;
typedef __attribute__((ext_vector_type(8)))  float    v8f;

#define N_ATOMS   100000
#define M_NBR     12
#define ATOM_F    64
#define NBR_F     41
#define IN_DIM    169      // 2*ATOM_F + NBR_F
#define KPAD      192      // 6 k-tiles of 32
#define OUT_DIM   128
#define ROWS_TOTAL (N_ATOMS * M_NBR)   // 1,200,000
#define XS_STRIDE 200      // f16 units; padded so 16-row frag loads hit 16 banks
#define GS_STRIDE 132      // f32 units; padded gated buffer stride

__device__ __forceinline__ float sigmoid_f(float x)  { return 1.f / (1.f + expf(-x)); }
__device__ __forceinline__ float softplus_f(float x) { return x > 20.f ? x : log1pf(expf(x)); }

// A fragment: 16x32 f16, lane m = l&15; VGPR j holds K pair per ISA table
// (j<4: K=2j, else K=16+2(j-4); +8 for lanes 16-31)
__device__ __forceinline__ v16h load_a_frag(const _Float16* Xs, int rowBase, int k0, int lane) {
  int m  = lane & 15;
  int kb = (lane & 16) ? 8 : 0;
  const uint32_t* p = reinterpret_cast<const uint32_t*>(Xs + (rowBase + m) * XS_STRIDE + k0);
  union { v16h h; uint32_t u[8]; } f;
#pragma unroll
  for (int j = 0; j < 8; ++j) {
    int koff = ((j < 4) ? (2 * j) : (16 + 2 * (j - 4))) + kb;
    f.u[j] = p[koff >> 1];
  }
  return f.h;
}

// B fragment: 32x16 f16, lane n = l&15; VGPR j holds K pair per ISA table (same as A)
// B[k][n] = W[colBase+n][k] (row-major f16 weights, stride KPAD, zero-padded)
__device__ __forceinline__ v16h load_b_frag(const _Float16* Wh, int colBase, int k0, int lane) {
  int n  = lane & 15;
  int kb = (lane & 16) ? 8 : 0;
  const uint32_t* p = reinterpret_cast<const uint32_t*>(Wh + (colBase + n) * KPAD + k0 + kb);
  union { v16h h; uint32_t u[8]; } f;
#pragma unroll
  for (int j = 0; j < 8; ++j) f.u[j] = p[(j < 4) ? j : (j + 4)];
  return f.h;
}

// Gather + concat + f32->f16 convert into LDS: row r -> (n=r/12, m=r%12),
// cols: [0,64)=atom_in[n], [64,128)=atom_in[idx[n,m]], [128,169)=nbr_fea, [169,192)=0
__device__ __forceinline__ void stage_rows(_Float16* Xs, int nRows, int rowG0,
                                           const float* __restrict__ atom_in,
                                           const float* __restrict__ nbr,
                                           const int*   __restrict__ idx) {
  for (int e = threadIdx.x; e < nRows * KPAD; e += 256) {
    int rl = e / KPAD, k = e - rl * KPAD;
    int r = rowG0 + rl;
    int n = r / M_NBR, m = r - n * M_NBR;
    float v;
    if (k < ATOM_F)          v = atom_in[n * ATOM_F + k];
    else if (k < 2 * ATOM_F) v = atom_in[idx[n * M_NBR + m] * ATOM_F + (k - ATOM_F)];
    else if (k < IN_DIM)     v = nbr[(n * M_NBR + m) * NBR_F + (k - 2 * ATOM_F)];
    else                     v = 0.f;
    Xs[rl * XS_STRIDE + k] = (_Float16)v;
  }
}

// Pass 0: weights f32 -> padded f16; zero the stat accumulators.
__global__ void prep_kernel(const float* __restrict__ fc_w, _Float16* __restrict__ Wh,
                            float* __restrict__ accums) {
  int e = blockIdx.x * 256 + threadIdx.x;
  if (e < OUT_DIM * KPAD) {
    int o = e / KPAD, k = e - o * KPAD;
    Wh[e] = (_Float16)((k < IN_DIM) ? fc_w[o * IN_DIM + k] : 0.f);
  }
  if (blockIdx.x == 0) {
    for (int i = threadIdx.x; i < 384; i += 256) accums[i] = 0.f;  // bn1 sum/sq + bn2 sum/sq
  }
}

// Pass 1: WMMA GEMM over 128-row tiles; reduce column sum & sum^2 of (X·Wᵀ + b) for BN1.
__global__ __launch_bounds__(256) void stats_gemm_kernel(
    const float* __restrict__ atom_in, const float* __restrict__ nbr,
    const int* __restrict__ idx, const _Float16* __restrict__ Wh,
    const float* __restrict__ fc_b,
    float* __restrict__ bn1_sum, float* __restrict__ bn1_sumsq) {
  __shared__ char smem[128 * XS_STRIDE * 2];   // 51,200 B
  _Float16* Xs = reinterpret_cast<_Float16*>(smem);
  stage_rows(Xs, 128, blockIdx.x * 128, atom_in, nbr, idx);
  __syncthreads();

  int lane = threadIdx.x & 31, wave = threadIdx.x >> 5;
  int rowBase = wave * 16;
  v8f acc[8] = {};
#pragma unroll
  for (int kt = 0; kt < 6; ++kt) {
    v16h a = load_a_frag(Xs, rowBase, kt * 32, lane);
#pragma unroll
    for (int ct = 0; ct < 8; ++ct) {
      v16h b = load_b_frag(Wh, ct * 16, kt * 32, lane);
      acc[ct] = __builtin_amdgcn_wmma_f32_16x16x32_f16(false, a, false, b,
                                                       (short)0, acc[ct], false, false);
    }
  }
  int n16 = lane & 15;
  float cs[8], cq[8];
#pragma unroll
  for (int ct = 0; ct < 8; ++ct) {
    float bv = fc_b[ct * 16 + n16];
    float s = 0.f, q = 0.f;
#pragma unroll
    for (int v = 0; v < 8; ++v) { float y = acc[ct][v] + bv; s += y; q += y * y; }
    s += __shfl_xor(s, 16, 32);   // fold lanes 16-31 (other 8 rows, same col)
    q += __shfl_xor(q, 16, 32);
    cs[ct] = s; cq[ct] = q;
  }
  __syncthreads();                 // X tile dead; reuse LDS for column reduce
  float* ls = reinterpret_cast<float*>(smem);   // [0,128)=sum, [128,256)=sumsq
  ls[threadIdx.x] = 0.f;
  __syncthreads();
  if (lane < 16) {
#pragma unroll
    for (int ct = 0; ct < 8; ++ct) {
      atomicAdd(&ls[ct * 16 + n16], cs[ct]);
      atomicAdd(&ls[128 + ct * 16 + n16], cq[ct]);
    }
  }
  __syncthreads();
  if (threadIdx.x < 128) atomicAdd(&bn1_sum[threadIdx.x], ls[threadIdx.x]);
  else                   atomicAdd(&bn1_sumsq[threadIdx.x - 128], ls[threadIdx.x]);
}

// Small finalize: mean/var -> affine scale/shift (biased var, torch/jax default).
__global__ void bn_finalize_kernel(const float* __restrict__ sum, const float* __restrict__ sumsq,
                                   const float* __restrict__ gamma, const float* __restrict__ beta,
                                   float* __restrict__ scale, float* __restrict__ shift,
                                   int n, float count) {
  int i = threadIdx.x;
  if (i >= n) return;
  float mean = sum[i] / count;
  float var  = sumsq[i] / count - mean * mean;
  float r  = rsqrtf(var + 1e-5f);
  float sc = gamma[i] * r;
  scale[i] = sc;
  shift[i] = beta[i] - mean * sc;
}

// Pass 2: recompute GEMM per 8-atom (96-row) tile, apply BN1 affine (bias folded),
// gate = sigmoid(z[:64]) * softplus(z[64:]), sum over M=12 in LDS,
// write nbr_sumed and accumulate BN2 column stats.
__global__ __launch_bounds__(256) void conv_gemm_kernel(
    const float* __restrict__ atom_in, const float* __restrict__ nbr,
    const int* __restrict__ idx, const _Float16* __restrict__ Wh,
    const float* __restrict__ fc_b,
    const float* __restrict__ bn1_scale, const float* __restrict__ bn1_shift,
    float* __restrict__ nbr_sumed, float* __restrict__ bn2_sum, float* __restrict__ bn2_sumsq) {
  __shared__ char smem[51200];   // X tile (38,400 B) then reused: gated 96x132 f32 + 512 B stats
  _Float16* Xs = reinterpret_cast<_Float16*>(smem);
  int atom0 = blockIdx.x * 8;
  stage_rows(Xs, 96, atom0 * M_NBR, atom_in, nbr, idx);
  __syncthreads();

  int lane = threadIdx.x & 31, wave = threadIdx.x >> 5;
  int colBase = wave * 16;       // each wave owns one 16-col stripe
  v8f acc[6] = {};
#pragma unroll
  for (int kt = 0; kt < 6; ++kt) {
    v16h b = load_b_frag(Wh, colBase, kt * 32, lane);
#pragma unroll
    for (int rt = 0; rt < 6; ++rt) {
      v16h a = load_a_frag(Xs, rt * 16, kt * 32, lane);
      acc[rt] = __builtin_amdgcn_wmma_f32_16x16x32_f16(false, a, false, b,
                                                       (short)0, acc[rt], false, false);
    }
  }
  int n16 = lane & 15, hi8 = (lane & 16) ? 8 : 0;
  int col = colBase + n16;
  float sc = bn1_scale[col];
  float sh = fc_b[col] * sc + bn1_shift[col];   // fold bias into BN1 shift
  __syncthreads();               // X/W reads done; reuse LDS as gated buffer
  float* G  = reinterpret_cast<float*>(smem);          // 96 x GS_STRIDE f32
  float* st = reinterpret_cast<float*>(smem + 50688);  // 128 f32: bn2 sum/sumsq partials
#pragma unroll
  for (int rt = 0; rt < 6; ++rt)
#pragma unroll
    for (int v = 0; v < 8; ++v)
      G[(rt * 16 + v + hi8) * GS_STRIDE + col] = acc[rt][v] * sc + sh;
  if (threadIdx.x < 128) st[threadIdx.x] = 0.f;
  __syncthreads();

  for (int o = threadIdx.x; o < 8 * ATOM_F; o += 256) {   // 512 outputs, 2 per thread
    int a = o >> 6, f = o & 63;
    float s = 0.f;
#pragma unroll
    for (int m = 0; m < M_NBR; ++m) {
      const float* row = &G[(a * M_NBR + m) * GS_STRIDE];
      s += sigmoid_f(row[f]) * softplus_f(row[f + ATOM_F]);
    }
    nbr_sumed[(atom0 + a) * ATOM_F + f] = s;
    atomicAdd(&st[f], s);
    atomicAdd(&st[64 + f], s * s);
  }
  __syncthreads();
  if (threadIdx.x < 64)        atomicAdd(&bn2_sum[threadIdx.x], st[threadIdx.x]);
  else if (threadIdx.x < 128)  atomicAdd(&bn2_sumsq[threadIdx.x - 64], st[threadIdx.x]);
}

// Pass 3: out = softplus(atom_in + BN2(nbr_sumed))
__global__ void out_kernel(const float* __restrict__ atom_in, const float* __restrict__ nbr_sumed,
                           const float* __restrict__ sc, const float* __restrict__ sh,
                           float* __restrict__ out) {
  int i = blockIdx.x * 256 + threadIdx.x;   // exactly 6.4M elements
  int f = i & 63;
  out[i] = softplus_f(atom_in[i] + nbr_sumed[i] * sc[f] + sh[f]);
}

extern "C" void kernel_launch(void* const* d_in, const int* in_sizes, int n_in,
                              void* d_out, int out_size, void* d_ws, size_t ws_size,
                              hipStream_t stream) {
  const float* atom_in = (const float*)d_in[0];
  const float* nbr     = (const float*)d_in[1];
  const int*   idx     = (const int*)d_in[2];
  const float* fc_w    = (const float*)d_in[3];
  const float* fc_b    = (const float*)d_in[4];
  const float* bn1_g   = (const float*)d_in[5];
  const float* bn1_b   = (const float*)d_in[6];
  const float* bn2_g   = (const float*)d_in[7];
  const float* bn2_b   = (const float*)d_in[8];
  float* out = (float*)d_out;

  char* ws = (char*)d_ws;
  _Float16* Wh     = (_Float16*)(ws + 0);        // 128x192 f16 = 49,152 B
  float* accums    = (float*)(ws + 49152);       // 384 floats, zeroed each call
  float* bn1_sum   = accums;                     // [128]
  float* bn1_sumsq = accums + 128;               // [128]
  float* bn2_sum   = accums + 256;               // [64]
  float* bn2_sumsq = accums + 320;               // [64]
  float* bn1_scale = (float*)(ws + 50688);       // [128]
  float* bn1_shift = (float*)(ws + 51200);       // [128]
  float* bn2_scale = (float*)(ws + 51712);       // [64]
  float* bn2_shift = (float*)(ws + 51968);       // [64]
  float* nbr_sumed = (float*)(ws + 52224);       // 100000x64 f32 = 25.6 MB

  prep_kernel<<<96, 256, 0, stream>>>(fc_w, Wh, accums);
  stats_gemm_kernel<<<ROWS_TOTAL / 128, 256, 0, stream>>>(atom_in, nbr, idx, Wh, fc_b,
                                                          bn1_sum, bn1_sumsq);
  bn_finalize_kernel<<<1, 128, 0, stream>>>(bn1_sum, bn1_sumsq, bn1_g, bn1_b,
                                            bn1_scale, bn1_shift, 128, (float)ROWS_TOTAL);
  conv_gemm_kernel<<<N_ATOMS / 8, 256, 0, stream>>>(atom_in, nbr, idx, Wh, fc_b,
                                                    bn1_scale, bn1_shift,
                                                    nbr_sumed, bn2_sum, bn2_sumsq);
  bn_finalize_kernel<<<1, 64, 0, stream>>>(bn2_sum, bn2_sumsq, bn2_g, bn2_b,
                                           bn2_scale, bn2_shift, 64, (float)N_ATOMS);
  out_kernel<<<(N_ATOMS * ATOM_F) / 256, 256, 0, stream>>>(atom_in, nbr_sumed,
                                                           bn2_scale, bn2_shift, out);
}
